// FFNNTransducerViterbi_50594714747167
// MI455X (gfx1250) — matrix operand order repaired and checked by reading, not compiled, not run on hardware
//
#include <hip/hip_runtime.h>
#include <hip/hip_bf16.h>

// ---------------- problem constants ----------------
#define B_    32
#define T_    2000
#define V_    79
#define E_    256
#define D_    640
#define HJ_   640
#define K1_   512            // CTX_H * E
#define NROWS (B_ * T_)      // 64000
#define MTILE 64             // rows per workgroup
#define ACT_STRIDE 648       // 640 + 8 bf16 pad -> LDS bank-conflict-free A-frag loads
#define VPAD  80             // V rounded up to 16

typedef __attribute__((ext_vector_type(16))) __bf16 v16bf;
typedef __attribute__((ext_vector_type(8)))  float  v8f;

static_assert(sizeof(__bf16) == 2, "bf16 size");

static __device__ __forceinline__ __bf16 f2bf(float f) {
    // round-to-nearest-even fp32 -> bf16
    unsigned u = __builtin_bit_cast(unsigned, f);
    unsigned r = u + 0x7fffu + ((u >> 16) & 1u);
    unsigned short h = (unsigned short)(r >> 16);
    return __builtin_bit_cast(__bf16, h);
}

// 16-byte global -> LDS copy on the gfx1250 async-DMA path (ASYNCcnt tracked).
// ldsOff: wave-relative LDS byte offset (low 32 bits of a generic LDS pointer).
static __device__ __forceinline__ void async_copy_b128(unsigned ldsOff, const void* gptr) {
    asm volatile("global_load_async_to_lds_b128 %0, %1, off"
                 :: "v"(ldsOff), "v"(gptr) : "memory");
}
static __device__ __forceinline__ void wait_async0() {
    asm volatile("s_wait_asynccnt 0x0" ::: "memory");
}

// ---------------- history shift-register (sequential over T, parallel over B) ----
__global__ void hist_kernel(const int* __restrict__ targets, int* __restrict__ hist) {
    int b = threadIdx.x;
    if (b >= B_) return;
    int h0 = 0, h1 = 0;                       // blank-initialized
    const int* tg = targets + (size_t)b * T_;
    int* hb = hist + (size_t)b * T_ * 2;
    for (int t = 0; t < T_; ++t) {
        int cur = tg[t];
        hb[2 * t]     = h0;                   // history BEFORE consuming targets[t]
        hb[2 * t + 1] = h1;
        if (cur != 0) { h1 = h0; h0 = cur; }  // shift only on non-blank
    }
}

// ---------------- embedding table -> bf16 (padding_idx row zeroed) -------------
__global__ void embcvt_kernel(const float* __restrict__ src, __bf16* __restrict__ dst) {
    int i = blockIdx.x * blockDim.x + threadIdx.x;
    if (i >= V_ * E_) return;
    float v = (i < E_) ? 0.f : src[i];        // row 0 (BLANK) zeroed
    dst[i] = f2bf(v);
}

// ---------------- weight [K][N] f32 -> transposed [Npad][K] bf16 ---------------
__global__ void wcvt_kernel(const float* __restrict__ src, __bf16* __restrict__ dst,
                            int K, int N, int Npad) {
    int i = blockIdx.x * blockDim.x + threadIdx.x;
    if (i >= Npad * K) return;
    int n = i / K, k = i - n * K;
    float v = (n < N) ? src[(size_t)k * N + n] : 0.f;
    dst[i] = f2bf(v);
}

// ---------------- one GEMM layer on a 64-row LDS tile --------------------------
// actIn LDS [MTILE][ACT_STRIDE] bf16; W global [Ntiles*16][K] bf16 (N-major);
// out: either bf16 relu activations to actOut, or raw f32 logits (stride VPAD).
static __device__ __forceinline__ void gemm_layer(
    const __bf16* __restrict__ actIn,
    __bf16*       __restrict__ actOut,
    float*        __restrict__ logitsOut,
    const __bf16* __restrict__ W,
    const float*  __restrict__ bias, int Nreal,
    int K, int Ntiles,
    const float*  __restrict__ srcAdd,      // joiner SUM input (stride D_) or null
    int lane, int mbase, int nhalf)
{
    const int lm = lane & 15;
    const int lh = lane >> 4;
    const __bf16* arow = actIn + (mbase + lm) * ACT_STRIDE;  // A: lane holds row M=lane&15
    for (int nt = nhalf; nt < Ntiles; nt += 2) {             // wave-uniform -> EXEC all 1s
        const int n = nt * 16 + lm;                          // B/C/D: lane holds col N=lane&15
        const __bf16* wrow = W + (size_t)n * K;
        v8f c = {};
        for (int k0 = 0; k0 < K; k0 += 32) {
            union { v16bf v; uint4 q[2]; } a, b;
            // A 16x32 bf16 layout: lane-half lh -> K = k0+8*lh+{0..7}, then +16
            const int ka = k0 + lh * 8;
            a.q[0] = *(const uint4*)(arow + ka);             // ds_load_b128
            a.q[1] = *(const uint4*)(arow + ka + 16);        // ds_load_b128
            // B 32x16 bf16 layout: lane-half lh -> K = k0+8*lh+{0..7}, then +16
            const int kb = k0 + lh * 8;
            b.q[0] = *(const uint4*)(wrow + kb);             // global_load_b128
            b.q[1] = *(const uint4*)(wrow + kb + 16);        // global_load_b128
            c = __builtin_amdgcn_wmma_f32_16x16x32_bf16(
                    false, a.v, false, b.v, (short)0, c, false, false);
        }
        const float bv = (n < Nreal) ? bias[n] : 0.f;
        if (logitsOut) {                                     // final layer: raw logits f32
            #pragma unroll
            for (int i = 0; i < 8; ++i) {
                int m = mbase + lh * 8 + i;                  // C VGPR i -> M = 8*lh + i
                logitsOut[m * VPAD + n] = c[i] + bv;
            }
        } else {
            #pragma unroll
            for (int i = 0; i < 8; ++i) {
                int m = mbase + lh * 8 + i;
                float v = c[i] + bv;
                v = v > 0.f ? v : 0.f;                       // ReLU
                if (srcAdd) v += srcAdd[(size_t)m * D_ + n]; // joiner SUM after ReLU
                actOut[m * ACT_STRIDE + n] = f2bf(v);
            }
        }
    }
}

// ---------------- fused predictor + joiner + log_softmax ----------------------
__global__ __launch_bounds__(256)
void fused_kernel(const int* __restrict__ hist,
                  const __bf16* __restrict__ emb,
                  const __bf16* __restrict__ wt1, const __bf16* __restrict__ wt2,
                  const __bf16* __restrict__ wt3, const __bf16* __restrict__ wt4,
                  const float* __restrict__ b1, const float* __restrict__ b2,
                  const float* __restrict__ b3, const float* __restrict__ b4,
                  const float* __restrict__ srcenc,
                  float* __restrict__ out)
{
    extern __shared__ char smem[];                           // 2 * 64*648 bf16 = 162 KB
    __bf16* actA = (__bf16*)smem;
    __bf16* actB = actA + MTILE * ACT_STRIDE;
    float*  logits = (float*)actA;                           // aliases actA (layer4 reads actB)
    float*  lse    = logits + MTILE * VPAD;                  // fits inside actA region

    const int tid   = threadIdx.x;
    const int lane  = tid & 31;
    const int wave  = tid >> 5;
    const int mbase = (wave & 3) * 16;                       // 4 waves over M
    const int nhalf = wave >> 2;                             // 2 waves over N tiles
    const int rowBase = blockIdx.x * MTILE;

    // Stage 0: gather concat(emb[h0], emb[h1]) -> actA[r][0..511] via async LDS-DMA.
    // Per-lane divergent gather (indexed by token history); no VGPR round-trip.
    {
        const unsigned aBase = (unsigned)(size_t)(void*)actA; // low 32 bits = LDS byte offset
        for (int v = tid; v < MTILE * (K1_ / 8); v += 256) {
            int r = v >> 6;                                  // 64 vec8 per row
            int e = (v & 63) * 8;                            // bf16 column 0..504
            int h = hist[(size_t)(rowBase + r) * 2 + (e >> 8)];  // e<256 -> h0 else h1
            const void* g = (const void*)(emb + (size_t)h * E_ + (e & (E_ - 1)));
            unsigned ldsOff = aBase + (unsigned)((r * ACT_STRIDE + e) * sizeof(__bf16));
            async_copy_b128(ldsOff, g);                      // global_load_async_to_lds_b128
        }
        wait_async0();                                       // s_wait_asynccnt 0 (own copies)
    }
    __syncthreads();                                         // everyone's copies visible

    // L1: relu(A0 @ pred_w1 + b1)         K=512 N=640
    gemm_layer(actA, actB, nullptr, wt1, b1, D_, K1_, D_ / 16, nullptr, lane, mbase, nhalf);
    __syncthreads();
    // L2: relu(. @ pred_w2 + b2) + src    K=640 N=640
    gemm_layer(actB, actA, nullptr, wt2, b2, D_, D_, D_ / 16,
               srcenc + (size_t)rowBase * D_, lane, mbase, nhalf);
    __syncthreads();
    // L3: relu(. @ join_w1 + jb1)         K=640 N=640
    gemm_layer(actA, actB, nullptr, wt3, b3, HJ_, D_, HJ_ / 16, nullptr, lane, mbase, nhalf);
    __syncthreads();
    // L4: logits = . @ join_w2 + jb2      K=640 N=80 (padded)
    gemm_layer(actB, nullptr, logits, wt4, b4, V_, HJ_, VPAD / 16, nullptr, lane, mbase, nhalf);
    __syncthreads();

    // log_softmax over V=79: per-row LSE then coalesced subtract+store
    if (tid < MTILE) {
        const float* lr = logits + tid * VPAD;
        float mx = lr[0];
        for (int v = 1; v < V_; ++v) mx = fmaxf(mx, lr[v]);
        float s = 0.f;
        for (int v = 0; v < V_; ++v) s += __expf(lr[v] - mx);
        lse[tid] = mx + __logf(s);
    }
    __syncthreads();
    for (int i = tid; i < MTILE * V_; i += 256) {
        int r = i / V_, v = i - r * V_;
        out[(size_t)(rowBase + r) * V_ + v] = logits[r * VPAD + v] - lse[r];
    }
}

// ---------------- launch --------------------------------------------------------
extern "C" void kernel_launch(void* const* d_in, const int* in_sizes, int n_in,
                              void* d_out, int out_size, void* d_ws, size_t ws_size,
                              hipStream_t stream) {
    const int*   targets = (const int*)d_in[0];
    const int*   tgt_len = (const int*)d_in[1];
    const float* srcenc  = (const float*)d_in[2];
    const int*   src_len = (const int*)d_in[3];
    const float* emb_f   = (const float*)d_in[4];
    const float* w1 = (const float*)d_in[5];
    const float* b1 = (const float*)d_in[6];
    const float* w2 = (const float*)d_in[7];
    const float* b2 = (const float*)d_in[8];
    const float* w3 = (const float*)d_in[9];
    const float* b3 = (const float*)d_in[10];
    const float* w4 = (const float*)d_in[11];
    const float* b4 = (const float*)d_in[12];

    // workspace layout (~2.95 MB, all 16B aligned)
    char* ws = (char*)d_ws;
    int*    hist = (int*)ws;     ws += (size_t)NROWS * 2 * sizeof(int);
    __bf16* emb  = (__bf16*)ws;  ws += (size_t)V_ * E_ * 2;
    __bf16* wt1  = (__bf16*)ws;  ws += (size_t)D_ * K1_ * 2;
    __bf16* wt2  = (__bf16*)ws;  ws += (size_t)D_ * D_ * 2;
    __bf16* wt3  = (__bf16*)ws;  ws += (size_t)HJ_ * D_ * 2;
    __bf16* wt4  = (__bf16*)ws;  ws += (size_t)VPAD * HJ_ * 2;

    hist_kernel<<<1, 32, 0, stream>>>(targets, hist);
    embcvt_kernel<<<(V_ * E_ + 255) / 256, 256, 0, stream>>>(emb_f, emb);
    wcvt_kernel<<<(D_ * K1_ + 255) / 256, 256, 0, stream>>>(w1, wt1, K1_, D_, D_);
    wcvt_kernel<<<(D_ * D_  + 255) / 256, 256, 0, stream>>>(w2, wt2, D_,  D_, D_);
    wcvt_kernel<<<(HJ_ * D_ + 255) / 256, 256, 0, stream>>>(w3, wt3, D_,  HJ_, HJ_);
    wcvt_kernel<<<(VPAD * HJ_ + 255) / 256, 256, 0, stream>>>(w4, wt4, HJ_, V_, VPAD);

    size_t shmem = (size_t)2 * MTILE * ACT_STRIDE * sizeof(__bf16);  // 165,888 B
    fused_kernel<<<NROWS / MTILE, 256, shmem, stream>>>(
        hist, emb, wt1, wt2, wt3, wt4, b1, b2, b3, b4, srcenc, (float*)d_out);

    // tuple tail: (out, source_lengths, target_lengths)
    float* outp = (float*)d_out;
    hipMemcpyAsync(outp + (size_t)NROWS * V_,      src_len, B_ * sizeof(int),
                   hipMemcpyDeviceToDevice, stream);
    hipMemcpyAsync(outp + (size_t)NROWS * V_ + B_, tgt_len, B_ * sizeof(int),
                   hipMemcpyDeviceToDevice, stream);
}
